// TruthGPT_Paper2510_00071_Integration_74131135529860
// MI455X (gfx1250) — compile-verified, matrix operand order checked by reading, not run on hardware
//
#include <hip/hip_runtime.h>

// Problem constants (from reference): B=4096, S=64, H=768
#define NB   4096
#define SEQ  64
#define HD   768
#define SH   (SEQ * HD)     // 49152 floats per item row
#define THR  0.85f

typedef __attribute__((ext_vector_type(16))) _Float16 v16h;
typedef __attribute__((ext_vector_type(8)))  _Float16 v8h;
typedef __attribute__((ext_vector_type(8)))  float    v8f;

// ---------------------------------------------------------------------------
// 1) Normalize last token of each item -> f16 matrix ln[NB][HD]
// ---------------------------------------------------------------------------
__global__ void k_normalize(const float* __restrict__ items,
                            _Float16* __restrict__ ln) {
    int b   = blockIdx.x;
    int tid = threadIdx.x;
    const float* row = items + (size_t)b * SH + (size_t)(SEQ - 1) * HD;

    float s = 0.f;
    for (int k = tid; k < HD; k += 256) { float v = row[k]; s += v * v; }

    __shared__ float red[256];
    red[tid] = s;
    __syncthreads();
    for (int off = 128; off > 0; off >>= 1) {
        if (tid < off) red[tid] += red[tid + off];
        __syncthreads();
    }
    float scale = 1.0f / (sqrtf(red[0]) + 1e-12f);
    for (int k = tid; k < HD; k += 256)
        ln[(size_t)b * HD + k] = (_Float16)(row[k] * scale);
}

// ---------------------------------------------------------------------------
// 2) sim = ln @ ln^T via v_wmma_f32_16x16x32_f16
//    Per ISA 16-bit A layout (16x32): lanes 0-15 hold K=0..7 and 16..23,
//    lanes 16-31 hold K=8..15 and 24..31, for row M = lane&15.
//    B = ln^T, so the B fragment for output column n is the same pattern
//    read from row n of ln.
// ---------------------------------------------------------------------------
__device__ inline v16h load_frag(const _Float16* __restrict__ ln,
                                 int row, int kb, int lane) {
    int koff = (lane < 16) ? 0 : 8;
    const _Float16* p = ln + (size_t)row * HD + kb + koff;
    v8h lo = *(const v8h*)(p);        // K = koff .. koff+7
    v8h hi = *(const v8h*)(p + 16);   // K = koff+16 .. koff+23
    v16h r;
#pragma unroll
    for (int i = 0; i < 8; ++i) { r[i] = lo[i]; r[i + 8] = hi[i]; }
    return r;
}

__global__ void k_sim_wmma(const _Float16* __restrict__ ln,
                           float* __restrict__ sim) {
    int tid  = threadIdx.x;
    int lane = tid & 31;
    int wave = tid >> 5;                      // 8 waves per block
    int m0   = blockIdx.y * 128 + wave * 16;  // 16 rows per wave
    int n0   = blockIdx.x * 64;               // 64 cols per wave
    int rowA = m0 + (lane & 15);
    int colL = lane & 15;

    v8f c0 = {}, c1 = {}, c2 = {}, c3 = {};

    for (int kb = 0; kb < HD; kb += 32) {
        if (kb + 32 < HD) {
            __builtin_prefetch(ln + (size_t)rowA * HD + kb + 32, 0, 1);
            __builtin_prefetch(ln + (size_t)(n0 + colL) * HD + kb + 32, 0, 1);
        }
        v16h a  = load_frag(ln, rowA,           kb, lane);
        v16h b0 = load_frag(ln, n0 + colL,      kb, lane);
        v16h b1 = load_frag(ln, n0 + 16 + colL, kb, lane);
        v16h b2 = load_frag(ln, n0 + 32 + colL, kb, lane);
        v16h b3 = load_frag(ln, n0 + 48 + colL, kb, lane);

        c0 = __builtin_amdgcn_wmma_f32_16x16x32_f16(false, a, false, b0,
                                                    (short)0, c0, false, false);
        c1 = __builtin_amdgcn_wmma_f32_16x16x32_f16(false, a, false, b1,
                                                    (short)0, c1, false, false);
        c2 = __builtin_amdgcn_wmma_f32_16x16x32_f16(false, a, false, b2,
                                                    (short)0, c2, false, false);
        c3 = __builtin_amdgcn_wmma_f32_16x16x32_f16(false, a, false, b3,
                                                    (short)0, c3, false, false);
    }

    // C/D layout: VGPR r holds M = r (lanes 0-15) and M = r+8 (lanes 16-31),
    // N = lane & 15.
    int rowoff = (lane >> 4) * 8;
#pragma unroll
    for (int r = 0; r < 8; ++r) {
        size_t base = (size_t)(m0 + rowoff + r) * NB + n0 + colL;
        sim[base +  0] = c0[r];
        sim[base + 16] = c1[r];
        sim[base + 32] = c2[r];
        sim[base + 48] = c3[r];
    }
}

// ---------------------------------------------------------------------------
// 3) Greedy first-unvisited-head clustering (sequential over i, parallel over j)
// ---------------------------------------------------------------------------
__global__ void k_cluster(const float* __restrict__ sim,
                          int* __restrict__ assigned_g) {
    __shared__ int asg[NB];
    int tid = threadIdx.x;
    for (int j = tid; j < NB; j += blockDim.x) asg[j] = -1;
    __syncthreads();

    for (int i = 0; i < NB; ++i) {
        int v = asg[i];
        __syncthreads();
        if (v < 0) {
            if (tid == 0) asg[i] = i;
            const float* row = sim + (size_t)i * NB;
            for (int j = i + 1 + tid; j < NB; j += blockDim.x)
                if (asg[j] < 0 && row[j] >= THR) asg[j] = i;
        }
        __syncthreads();
    }
    for (int j = tid; j < NB; j += blockDim.x) assigned_g[j] = asg[j];
}

// ---------------------------------------------------------------------------
// 4) Per-head centroid (mean of member flat rows), written into d_out region
// ---------------------------------------------------------------------------
__global__ void k_centroid(const float* __restrict__ items,
                           const int* __restrict__ assigned,
                           float* __restrict__ centroid) {
    int h = blockIdx.x;
    if (assigned[h] != h) return;   // only cluster heads (uniform branch)

    __shared__ int members[NB];
    __shared__ int cnt;
    int tid = threadIdx.x;
    if (tid == 0) cnt = 0;
    __syncthreads();
    for (int m = tid; m < NB; m += 256)
        if (assigned[m] == h) { int p = atomicAdd(&cnt, 1); members[p] = m; }
    __syncthreads();

    int n = cnt;
    float inv = 1.0f / (float)n;
    for (int k = tid; k < SH; k += 256) {
        float acc = 0.f;
        for (int idx = 0; idx < n; ++idx)
            acc += items[(size_t)members[idx] * SH + k];
        centroid[(size_t)h * SH + k] = acc * inv;
    }
}

// ---------------------------------------------------------------------------
// 5) dist[j] = || items[j] - centroid[assigned[j]] ||^2
// ---------------------------------------------------------------------------
__global__ void k_dist(const float* __restrict__ items,
                       const int* __restrict__ assigned,
                       const float* __restrict__ centroid,
                       float* __restrict__ dist) {
    int j   = blockIdx.x;
    int tid = threadIdx.x;
    int c   = assigned[j];
    const float* x  = items    + (size_t)j * SH;
    const float* mu = centroid + (size_t)c * SH;

    float s = 0.f;
    for (int k = tid; k < SH; k += 256) { float d = x[k] - mu[k]; s += d * d; }

    __shared__ float red[256];
    red[tid] = s;
    __syncthreads();
    for (int off = 128; off > 0; off >>= 1) {
        if (tid < off) red[tid] += red[tid + off];
        __syncthreads();
    }
    if (tid == 0) dist[j] = red[0];
}

// ---------------------------------------------------------------------------
// 6) best[h] = argmin over members of cluster h (ties -> smallest index)
// ---------------------------------------------------------------------------
__global__ void k_best(const int* __restrict__ assigned,
                       const float* __restrict__ dist,
                       int* __restrict__ best) {
    int h   = blockIdx.x;
    int tid = threadIdx.x;
    if (assigned[h] != h) { if (tid == 0) best[h] = h; return; }

    float bd = 3.4e38f;
    int   bi = 0x7FFFFFFF;
    for (int j = tid; j < NB; j += 256) {
        if (assigned[j] == h) {
            float d = dist[j];
            if (d < bd || (d == bd && j < bi)) { bd = d; bi = j; }
        }
    }
    __shared__ float sd[256];
    __shared__ int   si[256];
    sd[tid] = bd; si[tid] = bi;
    __syncthreads();
    for (int off = 128; off > 0; off >>= 1) {
        if (tid < off) {
            float od = sd[tid + off]; int oi = si[tid + off];
            if (od < sd[tid] || (od == sd[tid] && oi < si[tid])) {
                sd[tid] = od; si[tid] = oi;
            }
        }
        __syncthreads();
    }
    if (tid == 0) best[h] = si[0];
}

// ---------------------------------------------------------------------------
// 7) rep[j] = items[best[assigned[j]]]  (float4 streaming copy)
// ---------------------------------------------------------------------------
__global__ void k_rep(const float* __restrict__ items,
                      const int* __restrict__ assigned,
                      const int* __restrict__ best,
                      float* __restrict__ rep) {
    int j   = blockIdx.x;
    int src = best[assigned[j]];
    const float4* s = (const float4*)(items + (size_t)src * SH);
    float4*       d = (float4*)(rep + (size_t)j * SH);
    int k = blockIdx.y * 256 + threadIdx.x;   // SH/4 = 12288 float4 per row
    d[k] = s[k];
}

// ---------------------------------------------------------------------------
// 8) assigned (as float) + head_mask outputs
// ---------------------------------------------------------------------------
__global__ void k_meta(const int* __restrict__ assigned,
                       float* __restrict__ out_assigned,
                       float* __restrict__ out_mask) {
    int j = blockIdx.x * 256 + threadIdx.x;
    if (j < NB) {
        int a = assigned[j];
        out_assigned[j] = (float)a;
        out_mask[j]     = (a == j) ? 1.0f : 0.0f;
    }
}

// ---------------------------------------------------------------------------
extern "C" void kernel_launch(void* const* d_in, const int* in_sizes, int n_in,
                              void* d_out, int out_size, void* d_ws, size_t ws_size,
                              hipStream_t stream) {
    (void)in_sizes; (void)n_in; (void)out_size; (void)ws_size;

    const float* items = (const float*)d_in[0];
    float* out = (float*)d_out;

    // d_out layout: rep[NB*SH] | assigned[NB] | head_mask[NB]
    // rep region doubles as scratch: sim (first 64MB) then centroid (full).
    float* sim      = out;
    float* centroid = out;
    float* out_asg  = out + (size_t)NB * SH;
    float* out_mask = out_asg + NB;

    // d_ws layout: ln f16 (6 MB) | assigned int | dist f32 | best int
    _Float16* ln  = (_Float16*)d_ws;
    char* wsb     = (char*)d_ws + (size_t)NB * HD * sizeof(_Float16);
    int*   assigned = (int*)wsb;
    float* dist     = (float*)(assigned + NB);
    int*   best     = (int*)(dist + NB);

    // 1) normalize last tokens -> f16
    k_normalize<<<NB, 256, 0, stream>>>(items, ln);

    // 2) sim = ln @ ln^T (WMMA). Block = 8 waves covering 128x64 of sim.
    k_sim_wmma<<<dim3(NB / 64, NB / 128), 256, 0, stream>>>(ln, sim);

    // 3) greedy clustering (single workgroup, sequential heads)
    k_cluster<<<1, 1024, 0, stream>>>(sim, assigned);

    // 4) centroids (sim is dead; reuse d_out rep region)
    k_centroid<<<NB, 256, 0, stream>>>(items, assigned, centroid);

    // 5) distances to own centroid
    k_dist<<<NB, 256, 0, stream>>>(items, assigned, centroid, dist);

    // 6) per-cluster argmin
    k_best<<<NB, 256, 0, stream>>>(assigned, dist, best);

    // 7) representative gather (overwrites centroid scratch with final rep)
    k_rep<<<dim3(NB, SH / (4 * 256)), 256, 0, stream>>>(items, assigned, best, out);

    // 8) assigned + head_mask
    k_meta<<<NB / 256, 256, 0, stream>>>(assigned, out_asg, out_mask);
}
